// InvNet_49503793054498
// MI455X (gfx1250) — matrix operand order repaired and checked by reading, not compiled, not run on hardware
//
#include <hip/hip_runtime.h>
#include <hip/hip_bf16.h>

// ---------------------------------------------------------------------------
// Sizes from the reference
// ---------------------------------------------------------------------------
#define BDIM 256
#define FDIM 2048
#define CDIM 32768
#define PDIM 1041
#define SX   (CDIM + 1)        // 32769 logits per row (col0 = l_pos)
#define INV_BETA 20.0f         // 1/0.05

#define GEMM_BM 64
#define GEMM_BN 256
#define GEMM_BK 32

typedef __attribute__((ext_vector_type(16))) __bf16 v16bf;
typedef __attribute__((ext_vector_type(8)))  float  v8f;
typedef __attribute__((ext_vector_type(4)))  unsigned int v4u;
typedef __attribute__((ext_vector_type(4)))  float  v4f;

union Frag16 { v16bf v; v4u u[2]; };

// fp32 -> bf16 round-to-nearest-even, and back (manual, codegen-safe)
static __device__ __forceinline__ __bf16 f2bf(float f) {
    unsigned u = __builtin_bit_cast(unsigned, f);
    unsigned r = (u + 0x7FFFu + ((u >> 16) & 1u)) >> 16;
    unsigned short s = (unsigned short)r;
    return __builtin_bit_cast(__bf16, s);
}
static __device__ __forceinline__ float bf2f(__bf16 h) {
    unsigned short s = __builtin_bit_cast(unsigned short, h);
    unsigned u = ((unsigned)s) << 16;
    return __builtin_bit_cast(float, u);
}

// ---------------------------------------------------------------------------
// Kernel 1: row-normalize (f, ema, prob, prob_ema); for f also emit bf16 hi/lo
// grid (256, 4), block 256
// ---------------------------------------------------------------------------
__global__ void prep_normalize(const float* __restrict__ f,
                               const float* __restrict__ ema,
                               const float* __restrict__ p,
                               const float* __restrict__ pe,
                               float* __restrict__ fn, float* __restrict__ eman,
                               float* __restrict__ pn, float* __restrict__ pen,
                               __bf16* __restrict__ ahi, __bf16* __restrict__ alo) {
    const int b = blockIdx.x;
    const int which = blockIdx.y;
    const int t = threadIdx.x;

    const float* src; float* dst; int len;
    if (which == 0)      { src = f;   dst = fn;   len = FDIM; }
    else if (which == 1) { src = ema; dst = eman; len = FDIM; }
    else if (which == 2) { src = p;   dst = pn;   len = PDIM; }
    else                 { src = pe;  dst = pen;  len = PDIM; }
    src += (size_t)b * len;
    dst += (size_t)b * len;

    float s = 0.0f;
    for (int i = t; i < len; i += 256) { float v = src[i]; s += v * v; }
    __shared__ float red[256];
    red[t] = s; __syncthreads();
    for (int k = 128; k > 0; k >>= 1) {
        if (t < k) red[t] += red[t + k];
        __syncthreads();
    }
    const float inv = 1.0f / fmaxf(sqrtf(red[0]), 1e-12f);

    for (int i = t; i < len; i += 256) {
        float v = src[i] * inv;
        dst[i] = v;
        if (which == 0) {
            __bf16 h = f2bf(v);
            ahi[(size_t)b * FDIM + i] = h;
            alo[(size_t)b * FDIM + i] = f2bf(v - bf2f(h));
        }
    }
}

// ---------------------------------------------------------------------------
// Kernel 2: l_pos -> x[b][0] = dot(fn[b], eman[b]) / BETA.  grid 256, block 256
// ---------------------------------------------------------------------------
__global__ void lpos_kernel(const float* __restrict__ fn,
                            const float* __restrict__ eman,
                            float* __restrict__ xoh) {
    const int b = blockIdx.x, t = threadIdx.x;
    float s = 0.0f;
    for (int i = t; i < FDIM; i += 256)
        s += fn[(size_t)b * FDIM + i] * eman[(size_t)b * FDIM + i];
    __shared__ float red[256];
    red[t] = s; __syncthreads();
    for (int k = 128; k > 0; k >>= 1) {
        if (t < k) red[t] += red[t + k];
        __syncthreads();
    }
    if (t == 0) xoh[(size_t)b * SX] = red[0] * INV_BETA;
}

// ---------------------------------------------------------------------------
// Kernel 3: column scatter into the copied queues.  grid (256, 4), block 256
// ---------------------------------------------------------------------------
__global__ void scatter_cols(const float* __restrict__ fn,
                             const float* __restrict__ eman,
                             const float* __restrict__ pn,
                             const float* __restrict__ pen,
                             const int* __restrict__ targets,
                             float* __restrict__ q,  float* __restrict__ q2,
                             float* __restrict__ q3, float* __restrict__ q4) {
    const int b = blockIdx.x, w = blockIdx.y, t = threadIdx.x;
    const int tg = targets[b];
    const float* src; float* dst; int rows;
    if (w == 0)      { src = eman; dst = q;  rows = FDIM; }
    else if (w == 1) { src = fn;   dst = q2; rows = FDIM; }
    else if (w == 2) { src = pn;   dst = q3; rows = PDIM; }
    else             { src = pen;  dst = q4; rows = PDIM; }
    for (int r = t; r < rows; r += 256)
        dst[(size_t)r * CDIM + tg] = src[(size_t)b * rows + r];
}

// ---------------------------------------------------------------------------
// Kernel 4: l_neg GEMM, split-bf16 (hi/lo) with V_WMMA_F32_16X16X32_BF16.
// x[b][1+c] = (fn @ queue)[b][c] / BETA.   grid (128, 4), block 256 (8 waves)
// ---------------------------------------------------------------------------
__global__ __launch_bounds__(256) void gemm_kernel(
        const __bf16* __restrict__ Ahi, const __bf16* __restrict__ Alo,
        const float* __restrict__ Bq,   float* __restrict__ xoh) {
    __shared__ __align__(16) __bf16 sAh[GEMM_BM * GEMM_BK];
    __shared__ __align__(16) __bf16 sAl[GEMM_BM * GEMM_BK];
    __shared__ __align__(16) __bf16 sBh[GEMM_BN * GEMM_BK];   // K-major: [n][k]
    __shared__ __align__(16) __bf16 sBl[GEMM_BN * GEMM_BK];

    const int t    = threadIdx.x;
    const int n0   = blockIdx.x * GEMM_BN;
    const int m0   = blockIdx.y * GEMM_BM;
    const int wave = t >> 5, lane = t & 31;
    const int wm   = wave & 3;       // m-tile of 16 within the 64-row block
    const int wn   = wave >> 2;      // 0..1 -> n-tiles wn*8 .. wn*8+7
    const int lr   = lane & 15;
    const int lh   = lane >> 4;      // half-wave selector

    v8f acc[8];
    for (int i = 0; i < 8; i++)
        for (int j = 0; j < 8; j++) acc[i][j] = 0.0f;

    const int arow = t >> 2;          // 0..63
    const int acol = (t & 3) * 8;     // 0,8,16,24

    for (int kb = 0; kb < FDIM; kb += GEMM_BK) {
        // --- stage A tile (bf16 hi/lo, row-major [64][32]) ---
        *(v4u*)&sAh[arow * GEMM_BK + acol] =
            *(const v4u*)&Ahi[(size_t)(m0 + arow) * FDIM + kb + acol];
        *(v4u*)&sAl[arow * GEMM_BK + acol] =
            *(const v4u*)&Alo[(size_t)(m0 + arow) * FDIM + kb + acol];

        // --- stage B tile: fp32 global -> bf16 hi/lo, transposed to K-major ---
        for (int i = 0; i < 8; i++) {
            int qn = t * 8 + i;           // 0..2047 float4s
            int kk = qn >> 6;             // 0..31
            int n4 = qn & 63;             // 0..63
            v4f v = *(const v4f*)&Bq[(size_t)(kb + kk) * CDIM + n0 + n4 * 4];
            for (int j = 0; j < 4; j++) {
                float x = v[j];
                __bf16 h = f2bf(x);
                __bf16 l = f2bf(x - bf2f(h));
                int n = n4 * 4 + j;
                sBh[n * GEMM_BK + kk] = h;
                sBl[n * GEMM_BK + kk] = l;
            }
        }
        // prefetch next K-tile of queue (global_prefetch_b8)
        if (kb + GEMM_BK < FDIM) {
            int qn = t * 8;
            __builtin_prefetch(&Bq[(size_t)(kb + GEMM_BK + (qn >> 6)) * CDIM +
                                   n0 + (qn & 63) * 4], 0, 1);
        }
        __syncthreads();

        // --- A fragments (16x32 bf16, ISA lane/VGPR layout) ---
        Frag16 ah, al;
        const int am = wm * 16 + lr;
        ah.u[0] = *(const v4u*)&sAh[am * GEMM_BK + lh * 8];
        ah.u[1] = *(const v4u*)&sAh[am * GEMM_BK + 16 + lh * 8];
        al.u[0] = *(const v4u*)&sAl[am * GEMM_BK + lh * 8];
        al.u[1] = *(const v4u*)&sAl[am * GEMM_BK + 16 + lh * 8];

        for (int nt = 0; nt < 8; nt++) {
            const int bn = (wn * 8 + nt) * 16 + lr;
            Frag16 bh, bl;
            bh.u[0] = *(const v4u*)&sBh[bn * GEMM_BK + lh * 8];
            bh.u[1] = *(const v4u*)&sBh[bn * GEMM_BK + 16 + lh * 8];
            bl.u[0] = *(const v4u*)&sBl[bn * GEMM_BK + lh * 8];
            bl.u[1] = *(const v4u*)&sBl[bn * GEMM_BK + 16 + lh * 8];
            acc[nt] = __builtin_amdgcn_wmma_f32_16x16x32_bf16(
                false, ah.v, false, bh.v, (short)0, acc[nt], false, false);
            acc[nt] = __builtin_amdgcn_wmma_f32_16x16x32_bf16(
                false, ah.v, false, bl.v, (short)0, acc[nt], false, false);
            acc[nt] = __builtin_amdgcn_wmma_f32_16x16x32_bf16(
                false, al.v, false, bh.v, (short)0, acc[nt], false, false);
        }
        __syncthreads();
    }

    // epilogue: C/D layout -> x rows (scaled by 1/BETA), col offset +1 for l_pos
    const int rbase = m0 + wm * 16 + lh * 8;
    for (int nt = 0; nt < 8; nt++) {
        const int col = 1 + n0 + (wn * 8 + nt) * 16 + lr;
        for (int r = 0; r < 8; r++)
            xoh[(size_t)(rbase + r) * SX + col] = acc[nt][r] * INV_BETA;
    }
}

// ---------------------------------------------------------------------------
// Kernel 5: per-row top-7, log-softmax loss, dense smooth one-hot (in place).
// grid 256 (one block per row), block 256
// ---------------------------------------------------------------------------
__global__ void topk_softmax_kernel(float* __restrict__ xoh,
                                    const int* __restrict__ targets,
                                    float* __restrict__ rowloss) {
    const int b = blockIdx.x, t = threadIdx.x;
    float* xr = xoh + (size_t)b * SX;

    // thread-local top-7 (sorted descending)
    float lv[7]; int li[7];
    for (int j = 0; j < 7; j++) { lv[j] = -1e30f; li[j] = -1; }
    for (int i = t; i < SX; i += 256) {
        float v = xr[i];
        if (v > lv[6]) {
            int j = 6;
            while (j > 0 && v > lv[j - 1]) { lv[j] = lv[j - 1]; li[j] = li[j - 1]; j--; }
            lv[j] = v; li[j] = i;
        }
    }
    __shared__ float sv[256 * 7];
    __shared__ int   si[256 * 7];
    __shared__ float rv[256];
    __shared__ int   ri[256];
    __shared__ float topv[7];
    __shared__ int   topi[7];
    for (int j = 0; j < 7; j++) { sv[t * 7 + j] = lv[j]; si[t * 7 + j] = li[j]; }
    __syncthreads();

    // 7 rounds of block-parallel argmax with invalidation
    for (int k = 0; k < 7; k++) {
        float bv = -1e30f; int bp = t * 7;
        for (int j = 0; j < 7; j++) {
            float v = sv[t * 7 + j];
            if (v > bv) { bv = v; bp = t * 7 + j; }
        }
        rv[t] = bv; ri[t] = bp; __syncthreads();
        for (int s = 128; s > 0; s >>= 1) {
            if (t < s && rv[t + s] > rv[t]) { rv[t] = rv[t + s]; ri[t] = ri[t + s]; }
            __syncthreads();
        }
        if (t == 0) { topv[k] = rv[0]; topi[k] = si[ri[0]]; sv[ri[0]] = -1e30f; }
        __syncthreads();
    }

    // sum of exp(x - max), fixed-order tree reduce
    const float mx = topv[0];
    float s = 0.0f;
    for (int i = t; i < SX; i += 256) s += expf(xr[i] - mx);
    rv[t] = s; __syncthreads();
    for (int k = 128; k > 0; k >>= 1) {
        if (t < k) rv[t] += rv[t + k];
        __syncthreads();
    }
    const float logZ = logf(rv[0]);

    // thread 0: smooth weights + loss (needs x[target] before the row is zeroed)
    float wv[7]; int wi[7]; float xv[7]; int nw = 0;
    if (t == 0) {
        float e[6]; float se = 0.0f;
        const float m2 = topv[1];
        for (int j = 1; j < 7; j++) { e[j - 1] = expf(topv[j] - m2); se += e[j - 1]; }
        for (int j = 1; j < 7; j++) {
            wi[nw] = topi[j]; wv[nw] = e[j - 1] / se; xv[nw] = topv[j]; nw++;
        }
        const int tg = targets[b];
        const float xt = xr[tg];
        bool found = false;
        for (int i = 0; i < nw; i++)
            if (wi[i] == tg) { wv[i] = 1.0f; found = true; }
        if (!found) { wi[nw] = tg; wv[nw] = 1.0f; xv[nw] = xt; nw++; }
        float lr = 0.0f;
        for (int i = 0; i < nw; i++) lr += wv[i] * ((mx + logZ) - xv[i]);
        rowloss[b] = lr;
    }
    __syncthreads();

    // rewrite the row in place as the dense smooth one-hot
    for (int i = t; i < SX; i += 256) xr[i] = 0.0f;
    __syncthreads();
    if (t == 0)
        for (int i = 0; i < nw; i++) xr[wi[i]] = wv[i];
}

// ---------------------------------------------------------------------------
// Kernel 6: deterministic loss mean.  grid 1, block 256
// ---------------------------------------------------------------------------
__global__ void loss_reduce(const float* __restrict__ rowloss, float* __restrict__ out) {
    __shared__ float s[256];
    const int t = threadIdx.x;
    s[t] = rowloss[t]; __syncthreads();
    for (int k = 128; k > 0; k >>= 1) {
        if (t < k) s[t] += s[t + k];
        __syncthreads();
    }
    if (t == 0) out[0] = s[0] * (1.0f / 256.0f);
}

// ---------------------------------------------------------------------------
// Launch
// ---------------------------------------------------------------------------
extern "C" void kernel_launch(void* const* d_in, const int* in_sizes, int n_in,
                              void* d_out, int out_size, void* d_ws, size_t ws_size,
                              hipStream_t stream) {
    const float* f     = (const float*)d_in[0];
    const float* ema   = (const float*)d_in[1];
    const float* p     = (const float*)d_in[2];
    const float* pe    = (const float*)d_in[3];
    const float* queue  = (const float*)d_in[4];
    const float* queue2 = (const float*)d_in[5];
    const float* queue3 = (const float*)d_in[6];
    const float* queue4 = (const float*)d_in[7];
    const int*   targets = (const int*)d_in[8];

    // workspace layout
    const size_t FN_OFF   = 0;
    const size_t EMAN_OFF = FN_OFF   + (size_t)BDIM * FDIM;   // 524288
    const size_t PN_OFF   = EMAN_OFF + (size_t)BDIM * FDIM;
    const size_t PEN_OFF  = PN_OFF   + (size_t)BDIM * PDIM;
    const size_t RL_OFF   = PEN_OFF  + (size_t)BDIM * PDIM;
    const size_t F32_END  = RL_OFF + BDIM;                    // floats
    float* wsf = (float*)d_ws;
    float* fn   = wsf + FN_OFF;
    float* eman = wsf + EMAN_OFF;
    float* pn   = wsf + PN_OFF;
    float* pen  = wsf + PEN_OFF;
    float* rowloss = wsf + RL_OFF;
    __bf16* ahi = (__bf16*)((char*)d_ws + F32_END * sizeof(float));
    __bf16* alo = ahi + (size_t)BDIM * FDIM;

    // output layout: [loss, q, q2, q3, q4, oh]
    const size_t FC = (size_t)FDIM * CDIM;  // 67,108,864
    const size_t PC = (size_t)PDIM * CDIM;  // 34,111,488
    float* out = (float*)d_out;
    float* q   = out + 1;
    float* q2  = q  + FC;
    float* q3  = q2 + FC;
    float* q4  = q3 + PC;
    float* oh  = q4 + PC;                   // doubles as x = logits/BETA scratch

    prep_normalize<<<dim3(BDIM, 4), 256, 0, stream>>>(f, ema, p, pe,
                                                      fn, eman, pn, pen, ahi, alo);

    hipMemcpyAsync(q,  queue,  FC * sizeof(float), hipMemcpyDeviceToDevice, stream);
    hipMemcpyAsync(q2, queue2, FC * sizeof(float), hipMemcpyDeviceToDevice, stream);
    hipMemcpyAsync(q3, queue3, PC * sizeof(float), hipMemcpyDeviceToDevice, stream);
    hipMemcpyAsync(q4, queue4, PC * sizeof(float), hipMemcpyDeviceToDevice, stream);

    scatter_cols<<<dim3(BDIM, 4), 256, 0, stream>>>(fn, eman, pn, pen, targets,
                                                    q, q2, q3, q4);

    lpos_kernel<<<BDIM, 256, 0, stream>>>(fn, eman, oh);

    gemm_kernel<<<dim3(CDIM / GEMM_BN, BDIM / GEMM_BM), 256, 0, stream>>>(
        ahi, alo, queue, oh);

    topk_softmax_kernel<<<BDIM, 256, 0, stream>>>(oh, targets, rowloss);

    loss_reduce<<<1, 256, 0, stream>>>(rowloss, out);
}
